// NeuronInteraction_68341519614266
// MI455X (gfx1250) — matrix operand order repaired and checked by reading, not compiled, not run on hardware
//
// NeuronInteraction for MI455X (gfx1250, wave32, WMMA bf16 16x16x32).
// Pipeline: topk -> gather -> QKV GEMM -> WMMA attention -> out_proj(+scale)
//           -> su MLP (dense A-part + sparse folded msg-part) -> LayerNorm
//           -> au GEMM with fused gelu/dot epilogue -> sigmoid/clip -> topk -> scatter out.
#include <hip/hip_runtime.h>
#include <hip/hip_bf16.h>

typedef __attribute__((ext_vector_type(16))) __bf16 v16bf;
typedef __attribute__((ext_vector_type(8)))  float  v8f;
typedef __hip_bfloat16 bf16_t;

static constexpr int BATCH = 32;
static constexpr int NTOK  = 4096;
static constexpr int DIM   = 256;
static constexpr int KACT  = 256;   // MSG_K == sparsity_k == 256

#define WMMA_BF16(a, b, c) \
  __builtin_amdgcn_wmma_f32_16x16x32_bf16(false, (a), false, (b), (short)0, (c), false, false)

// ---------------- fragment loaders (per ISA 7.12.2) ----------------
// A 16x32 bf16: lane L holds row (L&15); K = {kb..kb+7, kb+16..kb+23}, kb = (L>>4)*8
static __device__ __forceinline__ v16bf load_fragA(const bf16_t* base, int ld) {
  const int lane = threadIdx.x & 31;
  const bf16_t* p = base + (size_t)(lane & 15) * ld + ((lane >> 4) << 3);
  union { v16bf v; uint4 q[2]; } u;
  u.q[0] = *(const uint4*)(p);
  u.q[1] = *(const uint4*)(p + 16);
  return u.v;
}
// B 32x16 bf16 (K-major source): lane L holds col (L&15); K = {kb..kb+7, kb+16..kb+23}, kb = (L>>4)*8
static __device__ __forceinline__ v16bf load_fragB(const bf16_t* base, int ld) {
  const int lane = threadIdx.x & 31;
  const bf16_t* p = base + (size_t)((lane >> 4) << 3) * ld + (lane & 15);
  union { v16bf v; bf16_t e[16]; } u;
  #pragma unroll
  for (int s = 0; s < 8; ++s) {
    u.e[s] = p[(size_t)s * ld];
    u.e[s + 8] = p[(size_t)(s + 16) * ld];
  }
  return u.v;
}

static __device__ __forceinline__ float redmax16(float v) {
  #pragma unroll
  for (int m = 1; m < 16; m <<= 1) v = fmaxf(v, __shfl_xor(v, m, 32));
  return v;
}
static __device__ __forceinline__ float redsum16(float v) {
  #pragma unroll
  for (int m = 1; m < 16; m <<= 1) v += __shfl_xor(v, m, 32);
  return v;
}
static __device__ __forceinline__ float gelu_f(float x) {
  return 0.5f * x * (1.0f + erff(x * 0.70710678118654752f));
}

// ---------------- elementwise / utility kernels ----------------
__global__ void cast_bf16_kernel(const float* __restrict__ in, bf16_t* __restrict__ out, int n) {
  int i = blockIdx.x * blockDim.x + threadIdx.x;
  if (i < n) out[i] = __float2bfloat16(in[i]);
}
// W [rows, cols] (out-major) -> Wt [cols, rows] (K-major) in bf16
__global__ void castT_bf16_kernel(const float* __restrict__ W, bf16_t* __restrict__ Wt,
                                  int rows, int cols) {
  int i = blockIdx.x * blockDim.x + threadIdx.x;
  if (i < rows * cols) {
    int r = i / cols, c = i % cols;
    Wt[(size_t)c * rows + r] = __float2bfloat16(W[(size_t)r * cols + c]);
  }
}
__global__ void gelu_cast_kernel(const float* __restrict__ in, bf16_t* __restrict__ out, int n) {
  int i = blockIdx.x * blockDim.x + threadIdx.x;
  if (i < n) out[i] = __float2bfloat16(gelu_f(in[i]));
}

// ---------------- per-batch exact top-256 via bitonic sort ----------------
__global__ __launch_bounds__(512) void topk_kernel(const float* __restrict__ act,
                                                   int* __restrict__ idx, float* __restrict__ vals) {
  __shared__ unsigned long long s[NTOK];
  const int b = blockIdx.x;
  const float* a = act + (size_t)b * NTOK;
  for (int i = threadIdx.x; i < NTOK; i += blockDim.x) {
    unsigned int bits = __float_as_uint(a[i]);           // values in [0,1] -> monotonic bits
    s[i] = ((unsigned long long)(bits ^ 0xFFFFFFFFu) << 32) | (unsigned int)i;
  }
  __syncthreads();
  for (int k = 2; k <= NTOK; k <<= 1) {
    for (int j = k >> 1; j > 0; j >>= 1) {
      for (int i = threadIdx.x; i < NTOK; i += blockDim.x) {
        int ixj = i ^ j;
        if (ixj > i) {
          unsigned long long x = s[i], y = s[ixj];
          bool up = ((i & k) == 0);
          if (up ? (x > y) : (x < y)) { s[i] = y; s[ixj] = x; }
        }
      }
      __syncthreads();
    }
  }
  for (int t = threadIdx.x; t < KACT; t += blockDim.x) {
    unsigned long long key = s[t];
    idx[b * KACT + t]  = (int)(unsigned int)key;
    vals[b * KACT + t] = __uint_as_float(~(unsigned int)(key >> 32));
  }
}

// gather active rows + cast to bf16: X[r,:] = hidden[b, idx[r], :]
__global__ __launch_bounds__(256) void gather_kernel(const float* __restrict__ hidden,
                                                     const int* __restrict__ idx,
                                                     bf16_t* __restrict__ X) {
  int r = blockIdx.x;
  int b = r >> 8;
  const float* src = hidden + ((size_t)b * NTOK + idx[r]) * DIM;
  X[(size_t)r * DIM + threadIdx.x] = __float2bfloat16(src[threadIdx.x]);
}

// ---------------- generic WMMA GEMM: one wave = 16 rows x 256 cols ----------------
enum { EPI_QKV = 0, EPI_OUTPROJ, EPI_SU1, EPI_MSGUP, EPI_SUW2, EPI_AU };

template <int EPI, bool TWO_A>
__global__ __launch_bounds__(256) void gemm_wmma(
    const bf16_t* __restrict__ A,   // [M,256] row-major bf16
    const bf16_t* __restrict__ A2,  // second A for K=512 (AU)
    const bf16_t* __restrict__ W,   // K-major weights [Kc, ldw]
    int ldw, int Kc,
    const float* __restrict__ bias,
    float* __restrict__ outF,
    bf16_t* __restrict__ outB, int ldo,
    const int* __restrict__ idx,        // MSGUP scatter indices
    const float* __restrict__ scalev,   // OUTPROJ per-row scale (topk vals)
    const float* __restrict__ w2,       // AU: au_w2 fp32 [256]
    float* __restrict__ delta) {        // AU: per-row dot output
  const int lane = threadIdx.x & 31;
  const int wave = threadIdx.x >> 5;
  const int m0 = (blockIdx.x * 8 + wave) * 16;
  const int n0 = blockIdx.y * 256;
  v8f acc[16];
  #pragma unroll
  for (int t = 0; t < 16; ++t) acc[t] = (v8f){0.f,0.f,0.f,0.f,0.f,0.f,0.f,0.f};

  for (int k0 = 0; k0 < Kc; k0 += 32) {
    const bf16_t* Ap = A; int kk = k0;
    if (TWO_A) { if (k0 >= 256) { Ap = A2; kk = k0 - 256; } }
    v16bf af = load_fragA(Ap + (size_t)m0 * 256 + kk, 256);
    #pragma unroll
    for (int t = 0; t < 16; ++t) {
      v16bf bfv = load_fragB(W + (size_t)k0 * ldw + n0 + t * 16, ldw);
      acc[t] = WMMA_BF16(af, bfv, acc[t]);
    }
  }

  const int nl = lane & 15;
  const int rh = (lane >> 4) << 3;   // C layout: lanes 0-15 -> rows 0-7, 16-31 -> rows 8-15
  if constexpr (EPI == EPI_AU) {
    float dp[8] = {0.f,0.f,0.f,0.f,0.f,0.f,0.f,0.f};
    #pragma unroll
    for (int t = 0; t < 16; ++t)
      #pragma unroll
      for (int r = 0; r < 8; ++r) {
        int col = n0 + t * 16 + nl;
        dp[r] += gelu_f(acc[t][r] + bias[col]) * w2[col];
      }
    #pragma unroll
    for (int r = 0; r < 8; ++r) {
      float s = redsum16(dp[r]);
      if (nl == 0) delta[m0 + rh + r] = s;   // wave owns full row: no atomics
    }
  } else {
    #pragma unroll
    for (int t = 0; t < 16; ++t)
      #pragma unroll
      for (int r = 0; r < 8; ++r) {
        int row = m0 + rh + r;
        int col = n0 + t * 16 + nl;
        float v = acc[t][r];
        if constexpr (EPI == EPI_QKV) {
          outB[(size_t)row * ldo + col] = __float2bfloat16(v + bias[col]);
        } else if constexpr (EPI == EPI_OUTPROJ) {
          outB[(size_t)row * ldo + col] = __float2bfloat16((v + bias[col]) * scalev[row]);
        } else if constexpr (EPI == EPI_SU1) {
          outF[(size_t)row * ldo + col] = v + bias[col];
        } else if constexpr (EPI == EPI_MSGUP) {
          int b = row >> 8;
          size_t g = ((size_t)b * NTOK + idx[row]) * ldo + col;
          outF[g] += v;                       // rows unique per batch: race-free
        } else { // EPI_SUW2
          outF[(size_t)row * ldo + col] = v + bias[col];
        }
      }
  }
}

// ---------------- WMMA attention: one block = (b,h,strip of 128 q rows) ----------------
__global__ __launch_bounds__(256) void attn_wmma(const bf16_t* __restrict__ qkv,
                                                 bf16_t* __restrict__ O) {
  __shared__ __align__(16) bf16_t smem[32768];      // 64KB: Kt then P (aliased)
  const int blk = blockIdx.x;
  const int bh = blk >> 1, strip = blk & 1;
  const int b = bh >> 2, h = bh & 3;
  const int row0 = b * KACT;
  // stage K^T [64, 256] (feature-major) into LDS
  bf16_t* Kt = smem;
  for (int t = threadIdx.x; t < KACT; t += 256) {
    const bf16_t* src = qkv + (size_t)(row0 + t) * 768 + 256 + h * 64;
    #pragma unroll
    for (int d = 0; d < 64; ++d) Kt[d * 256 + t] = src[d];
  }
  __syncthreads();

  const int wave = threadIdx.x >> 5, lane = threadIdx.x & 31;
  const int m0 = strip * 128 + wave * 16;
  const bf16_t* Qb = qkv + (size_t)(row0 + m0) * 768 + h * 64;
  v8f acc[16];
  #pragma unroll
  for (int t = 0; t < 16; ++t) acc[t] = (v8f){0.f,0.f,0.f,0.f,0.f,0.f,0.f,0.f};
  for (int k0 = 0; k0 < 64; k0 += 32) {
    v16bf af = load_fragA(Qb + k0, 768);
    #pragma unroll
    for (int t = 0; t < 16; ++t) {
      v16bf bfv = load_fragB(Kt + (size_t)k0 * 256 + t * 16, 256);
      acc[t] = WMMA_BF16(af, bfv, acc[t]);
    }
  }
  __syncthreads();   // done reading Kt; P aliases it

  // softmax over 256 cols (scale 1/sqrt(64)=0.125), write P bf16 to LDS in A-layout order
  const int nl = lane & 15, rh = (lane >> 4) << 3;
  bf16_t* P = smem + wave * (16 * 256);
  #pragma unroll
  for (int r = 0; r < 8; ++r) {
    float mx = -3.4e38f;
    #pragma unroll
    for (int t = 0; t < 16; ++t) mx = fmaxf(mx, acc[t][r]);
    mx = redmax16(mx) * 0.125f;
    float sm = 0.f;
    #pragma unroll
    for (int t = 0; t < 16; ++t) {
      float e = __expf(acc[t][r] * 0.125f - mx);
      acc[t][r] = e; sm += e;
    }
    float inv = 1.f / redsum16(sm);
    #pragma unroll
    for (int t = 0; t < 16; ++t)
      P[(rh + r) * 256 + t * 16 + nl] = __float2bfloat16(acc[t][r] * inv);
  }
  __syncthreads();

  // O = P @ V : V read directly from qkv (token-major == K-major for B frags)
  v8f oacc[4];
  #pragma unroll
  for (int t = 0; t < 4; ++t) oacc[t] = (v8f){0.f,0.f,0.f,0.f,0.f,0.f,0.f,0.f};
  const bf16_t* Vb = qkv + (size_t)row0 * 768 + 512 + h * 64;
  for (int k0 = 0; k0 < 256; k0 += 32) {
    v16bf af = load_fragA(P + k0, 256);
    #pragma unroll
    for (int t = 0; t < 4; ++t) {
      v16bf bfv = load_fragB(Vb + (size_t)k0 * 768 + t * 16, 768);
      oacc[t] = WMMA_BF16(af, bfv, oacc[t]);
    }
  }
  #pragma unroll
  for (int t = 0; t < 4; ++t)
    #pragma unroll
    for (int r = 0; r < 8; ++r) {
      int tok = m0 + rh + r;
      int col = h * 64 + t * 16 + nl;
      O[(size_t)(row0 + tok) * DIM + col] = __float2bfloat16(oacc[t][r]);
    }
}

// ---------------- LayerNorm: one wave per row of 256 ----------------
__global__ __launch_bounds__(256) void ln_kernel(const float* __restrict__ t,
                                                 const float* __restrict__ g,
                                                 const float* __restrict__ bta,
                                                 float* __restrict__ outF,
                                                 bf16_t* __restrict__ outB) {
  const int wave = threadIdx.x >> 5, lane = threadIdx.x & 31;
  const size_t row = (size_t)blockIdx.x * 8 + wave;
  const float* x = t + row * DIM;
  float v[8]; float s = 0.f;
  #pragma unroll
  for (int i = 0; i < 8; ++i) { v[i] = x[lane + 32 * i]; s += v[i]; }
  #pragma unroll
  for (int m = 1; m < 32; m <<= 1) s += __shfl_xor(s, m, 32);
  float mu = s * (1.f / DIM);
  float q = 0.f;
  #pragma unroll
  for (int i = 0; i < 8; ++i) { float d = v[i] - mu; q += d * d; }
  #pragma unroll
  for (int m = 1; m < 32; m <<= 1) q += __shfl_xor(q, m, 32);
  float inv = rsqrtf(q * (1.f / DIM) + 1e-5f);
  #pragma unroll
  for (int i = 0; i < 8; ++i) {
    int c = lane + 32 * i;
    float y = (v[i] - mu) * inv * g[c] + bta[c];
    outF[row * DIM + c] = y;
    outB[row * DIM + c] = __float2bfloat16(y);
  }
}

__global__ void actfin_kernel(const float* __restrict__ act, const float* __restrict__ delta,
                              const float* __restrict__ b2, float* __restrict__ newact, int n) {
  int i = blockIdx.x * blockDim.x + threadIdx.x;
  if (i < n) {
    float d = 1.f / (1.f + __expf(-(delta[i] + b2[0])));
    float na = 0.7f * act[i] + 0.3f * d;
    newact[i] = fminf(fmaxf(na, 0.f), 1.f);
  }
}

__global__ __launch_bounds__(256) void scatter_out_kernel(const int* __restrict__ idx2,
                                                          const float* __restrict__ vals2,
                                                          const float* __restrict__ newh,
                                                          float* __restrict__ outAct,
                                                          float* __restrict__ outHid) {
  int r = blockIdx.x;
  int b = r >> 8;
  size_t g = (size_t)b * NTOK + idx2[r];
  if (threadIdx.x == 0) outAct[g] = vals2[r];
  outHid[g * DIM + threadIdx.x] = newh[g * DIM + threadIdx.x];
}

// ---------------- host ----------------
extern "C" void kernel_launch(void* const* d_in, const int* in_sizes, int n_in,
                              void* d_out, int out_size, void* d_ws, size_t ws_size,
                              hipStream_t stream) {
  const float* act    = (const float*)d_in[0];
  const float* hidden = (const float*)d_in[1];
  const float* in_w   = (const float*)d_in[2];
  const float* in_b   = (const float*)d_in[3];
  const float* out_w  = (const float*)d_in[4];
  const float* out_b  = (const float*)d_in[5];
  const float* su_w1  = (const float*)d_in[6];
  const float* su_b1  = (const float*)d_in[7];
  const float* su_w2  = (const float*)d_in[8];
  const float* su_b2  = (const float*)d_in[9];
  const float* au_w1  = (const float*)d_in[10];
  const float* au_b1  = (const float*)d_in[11];
  const float* au_w2  = (const float*)d_in[12];
  const float* au_b2  = (const float*)d_in[13];
  const float* ln_g   = (const float*)d_in[14];
  const float* ln_b   = (const float*)d_in[15];

  const size_t BN  = (size_t)BATCH * NTOK;        // 131072
  const size_t BK  = (size_t)BATCH * KACT;        // 8192
  char* ws = (char*)d_ws;
  size_t off = 0;
  auto bump = [&](size_t bytes) { char* p = ws + off; off = (off + bytes + 255) & ~(size_t)255; return p; };

  bf16_t* hiddenB = (bf16_t*)bump(BN * DIM * 2);          // 64MB
  bf16_t* wInT    = (bf16_t*)bump((size_t)256 * 768 * 2);
  bf16_t* wOutT   = (bf16_t*)bump((size_t)256 * 256 * 2);
  bf16_t* wSu1T   = (bf16_t*)bump((size_t)512 * 256 * 2);
  bf16_t* wSu2T   = (bf16_t*)bump((size_t)256 * 256 * 2);
  bf16_t* wAu1T   = (bf16_t*)bump((size_t)512 * 256 * 2);
  int*    idx1    = (int*)bump(BK * 4);
  float*  vals1   = (float*)bump(BK * 4);
  bf16_t* XactB   = (bf16_t*)bump(BK * DIM * 2);          // 4MB
  bf16_t* qkvB    = (bf16_t*)bump(BK * 768 * 2);          // 12MB
  bf16_t* OB      = (bf16_t*)bump(BK * DIM * 2);          // 4MB
  bf16_t* msgB    = (bf16_t*)bump(BK * DIM * 2);          // 4MB
  float*  hlin    = (float*)bump(BN * DIM * 4);           // 128MB (reused as pre-LN t)
  bf16_t* hB      = (bf16_t*)bump(BN * DIM * 2);          // 64MB
  float*  newh    = (float*)bump(BN * DIM * 4);           // 128MB
  bf16_t* newhB   = (bf16_t*)bump(BN * DIM * 2);          // 64MB
  float*  delta   = (float*)bump(BN * 4);
  float*  newact  = (float*)bump(BN * 4);
  int*    idx2    = (int*)bump(BK * 4);
  float*  vals2   = (float*)bump(BK * 4);
  (void)ws_size; (void)n_in; (void)in_sizes; (void)out_size;

  // 1) casts: hidden -> bf16; weights -> K-major bf16
  cast_bf16_kernel<<<(int)((BN * DIM + 255) / 256), 256, 0, stream>>>(hidden, hiddenB, (int)(BN * DIM));
  castT_bf16_kernel<<<(768 * 256 + 255) / 256, 256, 0, stream>>>(in_w,  wInT,  768, 256);
  castT_bf16_kernel<<<(256 * 256 + 255) / 256, 256, 0, stream>>>(out_w, wOutT, 256, 256);
  castT_bf16_kernel<<<(256 * 512 + 255) / 256, 256, 0, stream>>>(su_w1, wSu1T, 256, 512);
  castT_bf16_kernel<<<(256 * 256 + 255) / 256, 256, 0, stream>>>(su_w2, wSu2T, 256, 256);
  castT_bf16_kernel<<<(256 * 512 + 255) / 256, 256, 0, stream>>>(au_w1, wAu1T, 256, 512);

  // 2) top-256 per batch, gather active states
  topk_kernel<<<BATCH, 512, 0, stream>>>(act, idx1, vals1);
  gather_kernel<<<(int)BK, 256, 0, stream>>>(hidden, idx1, XactB);

  // 3) sparse MHA
  gemm_wmma<EPI_QKV, false><<<dim3(64, 3), 256, 0, stream>>>(
      XactB, nullptr, wInT, 768, 256, in_b, nullptr, qkvB, 768, nullptr, nullptr, nullptr, nullptr);
  attn_wmma<<<BATCH * 4 * 2, 256, 0, stream>>>(qkvB, OB);
  gemm_wmma<EPI_OUTPROJ, false><<<dim3(64, 1), 256, 0, stream>>>(
      OB, nullptr, wOutT, 256, 256, out_b, nullptr, msgB, 256, nullptr, vals1, nullptr, nullptr);

  // 4) su MLP: dense hidden part + folded sparse message part, gelu, su_w2, LN
  gemm_wmma<EPI_SU1, false><<<dim3(1024, 1), 256, 0, stream>>>(
      hiddenB, nullptr, wSu1T, 256, 256, su_b1, hlin, nullptr, 256, nullptr, nullptr, nullptr, nullptr);
  gemm_wmma<EPI_MSGUP, false><<<dim3(64, 1), 256, 0, stream>>>(
      msgB, nullptr, wSu1T + (size_t)256 * 256, 256, 256, nullptr, hlin, nullptr, 256, idx1, nullptr, nullptr, nullptr);
  gelu_cast_kernel<<<(int)((BN * DIM + 255) / 256), 256, 0, stream>>>(hlin, hB, (int)(BN * DIM));
  gemm_wmma<EPI_SUW2, false><<<dim3(1024, 1), 256, 0, stream>>>(
      hB, nullptr, wSu2T, 256, 256, su_b2, hlin, nullptr, 256, nullptr, nullptr, nullptr, nullptr);
  ln_kernel<<<(int)(BN / 8), 256, 0, stream>>>(hlin, ln_g, ln_b, newh, newhB);

  // 5) au path: K=512 GEMM with fused gelu + au_w2 dot epilogue, then sigmoid/clip
  gemm_wmma<EPI_AU, true><<<dim3(1024, 1), 256, 0, stream>>>(
      hiddenB, newhB, wAu1T, 256, 512, au_b1, nullptr, nullptr, 256, nullptr, nullptr, au_w2, delta);
  actfin_kernel<<<(int)((BN + 255) / 256), 256, 0, stream>>>(act, delta, au_b2, newact, (int)BN);

  // 6) final top-k + sparse outputs
  topk_kernel<<<BATCH, 512, 0, stream>>>(newact, idx2, vals2);
  float* outAct = (float*)d_out;
  float* outHid = outAct + BN;
  hipMemsetAsync(d_out, 0, (BN + BN * DIM) * sizeof(float), stream);
  scatter_out_kernel<<<(int)BK, 256, 0, stream>>>(idx2, vals2, newh, outAct, outHid);
}
